// HyperGAT_27144193311253
// MI455X (gfx1250) — matrix operand order repaired and checked by reading, not compiled, not run on hardware
//
#include <hip/hip_runtime.h>

#define EPSF 1e-6f
#define NN   6144
#define DD   64

typedef __attribute__((ext_vector_type(16))) __bf16 v16bf;
typedef __attribute__((ext_vector_type(8)))  float  v8f;

union Frag {
    v16bf v;
    unsigned short u[16];
    uint4 q[2];
};

static __device__ __forceinline__ unsigned short f2bf(float f) {
    unsigned int u = __float_as_uint(f);
    unsigned int r = (u + 0x7FFFu + ((u >> 16) & 1u)) >> 16;   // RNE
    return (unsigned short)r;
}

static __device__ __forceinline__ float artanh_clamped(float x) {
    x = fminf(fmaxf(x, -1.f + EPSF), 1.f - EPSF);
    return 0.5f * __logf((1.f + x) / (1.f - x));
}

// ---------------------------------------------------------------------------
// Kernel 1: Mobius matvec P = mobius_matvec(W, H); also emit n2 = ||P||^2,
// P in bf16 (row major, N x 64) and Ht = H^T in bf16 (64 x N).
// One block (64 threads) per row i.
// ---------------------------------------------------------------------------
__global__ __launch_bounds__(64) void k_mobius(
    const float* __restrict__ Hin, const float* __restrict__ W,
    unsigned short* __restrict__ Pbf, float* __restrict__ n2,
    unsigned short* __restrict__ Htb) {
    const int i = blockIdx.x;
    const int j = threadIdx.x;            // 0..63
    __shared__ float hrow[DD];
    __shared__ float red[DD];

    float hv = Hin[i * DD + j];
    hrow[j] = hv;
    red[j] = hv * hv;
    __syncthreads();
    for (int s = 32; s > 0; s >>= 1) {
        if (j < s) red[j] += red[j + s];
        __syncthreads();
    }
    float xn2 = red[0];
    __syncthreads();

    float mx = 0.f;
#pragma unroll
    for (int k = 0; k < DD; ++k) mx += hrow[k] * W[j * DD + k];

    red[j] = mx * mx;
    __syncthreads();
    for (int s = 32; s > 0; s >>= 1) {
        if (j < s) red[j] += red[j + s];
        __syncthreads();
    }
    float mxn2 = red[0];

    float xn  = fmaxf(sqrtf(xn2),  EPSF);
    float mxn = fmaxf(sqrtf(mxn2), EPSF);
    float at  = artanh_clamped(xn);                    // sc = 1
    float scale = tanhf(mxn / xn * at) / mxn;

    Pbf[i * DD + j]   = f2bf(scale * mx);
    Htb[(size_t)j * NN + i] = f2bf(hv);
    if (j == 0) n2[i] = scale * scale * mxn2;          // ||P_i||^2
}

// ---------------------------------------------------------------------------
// Kernel 2: fused hyperbolic distance + masked online softmax + att @ H.
// One wave (32 threads) per 16-row output tile; streams 32 columns / iter.
// C-layout row for lane L, register r:  row = r + 8*(L>=16), col = L%16.
// ---------------------------------------------------------------------------
__global__ __launch_bounds__(32) void k_attn(
    const unsigned short* __restrict__ Pbf, const float* __restrict__ n2,
    const unsigned short* __restrict__ Htb, const int* __restrict__ Amask,
    float* __restrict__ Hout) {
    const int L  = threadIdx.x;
    const int lm = L & 15;
    const int lh = L >> 4;
    const int i0 = blockIdx.x * 16;

    __shared__ float plds[16 * 32];

    // A fragments of the wave's 16 P rows (K = 64 -> two K=32 frags).
    // 16-bit A layout: lane L, element t -> K = t + 8*(t>=8) + 8*(L>=16).
    Frag aP[2];
#pragma unroll
    for (int f = 0; f < 2; ++f) {
        const unsigned short* rp = Pbf + (i0 + lm) * DD + 32 * f + 8 * lh;
        aP[f].q[0] = *(const uint4*)(rp);        // t = 0..7
        aP[f].q[1] = *(const uint4*)(rp + 16);   // t = 8..15 (K += 16)
    }

    float ni[8];
#pragma unroll
    for (int r = 0; r < 8; ++r) ni[r] = n2[i0 + r + 8 * lh];

    v8f acc0 = {0,0,0,0,0,0,0,0}, acc1 = acc0, acc2 = acc0, acc3 = acc0;
    float m[8], l[8];
#pragma unroll
    for (int r = 0; r < 8; ++r) { m[r] = -3.0e38f; l[r] = 0.f; }

    for (int j0 = 0; j0 < NN; j0 += 32) {
        // ---- Gram tiles: s[h] = P_i(16x64) x P_{j0+16h..}^T (64x16) ----
        v8f s[2];
#pragma unroll
        for (int h = 0; h < 2; ++h) {
            // 16-bit B layout: lane L, element t -> (K = t + 8*(t>=8) + 8*(L>=16), N = L%16)
            const unsigned short* prow = Pbf + (j0 + 16 * h + lm) * DD + 8 * lh;
            Frag b0, b1;
            b0.q[0] = *(const uint4*)(prow);
            b0.q[1] = *(const uint4*)(prow + 16);
            b1.q[0] = *(const uint4*)(prow + 32);
            b1.q[1] = *(const uint4*)(prow + 48);
            v8f c = {0,0,0,0,0,0,0,0};
            c = __builtin_amdgcn_wmma_f32_16x16x32_bf16(false, aP[0].v, false, b0.v,
                                                        (short)0, c, false, false);
            c = __builtin_amdgcn_wmma_f32_16x16x32_bf16(false, aP[1].v, false, b1.v,
                                                        (short)0, c, false, false);
            s[h] = c;
        }

        // ---- hyperbolic squared distance + mask ----
        float sv[2][8];
        int   mk[2][8];
        float tmax[8];
#pragma unroll
        for (int r = 0; r < 8; ++r) tmax[r] = -3.0e38f;
#pragma unroll
        for (int h = 0; h < 2; ++h) {
            float nj = n2[j0 + 16 * h + lm];
#pragma unroll
            for (int r = 0; r < 8; ++r) {
                float xy = s[h][r];
                float a = 1.f - 2.f * xy + nj;
                float b = 1.f - ni[r];
                float denom = fmaxf(1.f - 2.f * xy + ni[r] * nj, EPSF);
                float num2 = a * a * ni[r] - 2.f * a * b * xy + b * b * nj;
                float mn = sqrtf(fmaxf(num2, 0.f)) / denom;
                float d = 2.f * artanh_clamped(mn);
                float val = d * d;
                int row = i0 + r + 8 * lh;
                int msk = Amask[(size_t)row * NN + j0 + 16 * h + lm] > 0;
                sv[h][r] = msk ? val : -1.0e4f;
                mk[h][r] = msk;
                tmax[r] = fmaxf(tmax[r], sv[h][r]);
            }
        }
        // row max across the 16-lane group
#pragma unroll
        for (int r = 0; r < 8; ++r) {
            float t = tmax[r];
            t = fmaxf(t, __shfl_xor(t, 1, 32));
            t = fmaxf(t, __shfl_xor(t, 2, 32));
            t = fmaxf(t, __shfl_xor(t, 4, 32));
            t = fmaxf(t, __shfl_xor(t, 8, 32));
            tmax[r] = t;
        }
        // ---- online softmax update; stage p through LDS ----
#pragma unroll
        for (int r = 0; r < 8; ++r) {
            float mnew  = fmaxf(m[r], tmax[r]);
            float alpha = __expf(m[r] - mnew);
            float psum = 0.f;
#pragma unroll
            for (int h = 0; h < 2; ++h) {
                float p = mk[h][r] ? __expf(sv[h][r] - mnew) : 0.f;
                plds[(r + 8 * lh) * 32 + 16 * h + lm] = p;
                psum += p;
            }
            psum += __shfl_xor(psum, 1, 32);
            psum += __shfl_xor(psum, 2, 32);
            psum += __shfl_xor(psum, 4, 32);
            psum += __shfl_xor(psum, 8, 32);
            l[r] = l[r] * alpha + psum;
            m[r] = mnew;
            acc0[r] *= alpha; acc1[r] *= alpha; acc2[r] *= alpha; acc3[r] *= alpha;
        }

        // ---- re-layout p (16x32) into a K=32 bf16 A fragment ----
        Frag aAtt;
        {
            const float* src0 = &plds[lm * 32 + 8 * lh];        // t=0..7  -> K = 8*lh + t
            const float* src1 = &plds[lm * 32 + 16 + 8 * lh];   // t=8..15 -> K = 16 + 8*lh + (t-8)
#pragma unroll
            for (int t = 0; t < 8; ++t) {
                aAtt.u[t]     = f2bf(src0[t]);
                aAtt.u[8 + t] = f2bf(src1[t]);
            }
        }

        // ---- acc += att(16x32) @ H_j(32x64) via 4 WMMAs over col blocks ----
#pragma unroll
        for (int cb = 0; cb < 4; ++cb) {
            const unsigned short* hrow =
                Htb + (size_t)(cb * 16 + lm) * NN + j0 + 8 * lh;
            Frag bH;
            bH.q[0] = *(const uint4*)(hrow);
            bH.q[1] = *(const uint4*)(hrow + 16);
            v8f& acc = (cb == 0) ? acc0 : (cb == 1) ? acc1 : (cb == 2) ? acc2 : acc3;
            acc = __builtin_amdgcn_wmma_f32_16x16x32_bf16(false, aAtt.v, false, bH.v,
                                                          (short)0, acc, false, false);
        }
    }

    // ---- normalize and write ----
#pragma unroll
    for (int r = 0; r < 8; ++r) {
        float inv = 1.f / fmaxf(l[r], 1e-30f);
        int row = i0 + r + 8 * lh;
        Hout[row * DD +  0 + lm] = acc0[r] * inv;
        Hout[row * DD + 16 + lm] = acc1[r] * inv;
        Hout[row * DD + 32 + lm] = acc2[r] * inv;
        Hout[row * DD + 48 + lm] = acc3[r] * inv;
    }
}

extern "C" void kernel_launch(void* const* d_in, const int* in_sizes, int n_in,
                              void* d_out, int out_size, void* d_ws, size_t ws_size,
                              hipStream_t stream) {
    const float* H = (const float*)d_in[0];   // N x 64 f32
    const float* W = (const float*)d_in[1];   // 2 x 64 x 64 f32
    const int*   A = (const int*)d_in[2];     // N x N int32
    float* out = (float*)d_out;               // N x 64 f32

    // workspace layout (all 256B-aligned offsets)
    unsigned short* Pbf = (unsigned short*)d_ws;                          // N*64*2 B
    float*          nn2 = (float*)((char*)d_ws + (size_t)NN * DD * 2);    // N*4   B
    unsigned short* Htb = (unsigned short*)((char*)d_ws + (size_t)NN * DD * 2
                                                        + (size_t)NN * 4); // 64*N*2 B

    for (int layer = 0; layer < 2; ++layer) {
        const float* Hin = (layer == 0) ? H : out;
        k_mobius<<<NN, 64, 0, stream>>>(Hin, W + layer * DD * DD, Pbf, nn2, Htb);
        k_attn<<<NN / 16, 32, 0, stream>>>(Pbf, nn2, Htb, A, out);
    }
}
